// KTModel_84679575208065
// MI455X (gfx1250) — matrix operand order repaired and checked by reading, not compiled, not run on hardware
//
#include <hip/hip_runtime.h>

typedef __attribute__((ext_vector_type(16))) _Float16 v16h;
typedef __attribute__((ext_vector_type(8)))  _Float16 v8h;
typedef __attribute__((ext_vector_type(8)))  float    v8f;
typedef __attribute__((ext_vector_type(4)))  float    v4f;
typedef __attribute__((ext_vector_type(4)))  unsigned int v4u;

#define DEV __device__ __forceinline__

constexpr int BS = 512, SEQL = 200, DIM = 128;

// workspace layout, in _Float16 elements
constexpr size_t SZ_XH    = (size_t)BS * SEQL * DIM;        // x activations, f16
constexpr size_t OFF_XH   = 0;
constexpr size_t OFF_FCW  = OFF_XH  + SZ_XH;                // fc_W  packed (256x128)
constexpr size_t OFF_FC2W = OFF_FCW + 256 * 128;            // fc2_W packed (256x128)
constexpr size_t OFF_WQ2  = OFF_FC2W + 256 * 128;
constexpr size_t OFF_WK2  = OFF_WQ2 + 128 * 128;
constexpr size_t OFF_WV2  = OFF_WK2 + 128 * 128;
constexpr size_t OFF_WQ3  = OFF_WV2 + 128 * 128;
constexpr size_t OFF_WK3  = OFF_WQ3 + 128 * 128;
constexpr size_t OFF_WV3  = OFF_WK3 + 128 * 128;
constexpr size_t OFF_F11  = OFF_WV3 + 128 * 128;
constexpr size_t OFF_F12  = OFF_F11 + 128 * 128;
constexpr size_t OFF_F21  = OFF_F12 + 128 * 128;
constexpr size_t OFF_F22  = OFF_F21 + 128 * 128;
constexpr size_t OFF_SKH  = OFF_F22 + 128 * 128;            // skill_emb f16 (1002x128)
constexpr size_t OFF_ANH  = OFF_SKH + 1002 * 128;           // answer_emb f16 (3x128)
constexpr size_t OFF_DFH  = OFF_ANH + 3 * 128;              // diff_emb f16 (102x128)
constexpr size_t OFF_HNH  = OFF_DFH + 102 * 128;            // hints_emb f16 (12x128)
constexpr size_t WS_HALVES = OFF_HNH + 12 * 128;

// scan kernel geometry: 8 blocks x 4 waves, 16 batch rows per wave
constexpr int SCAN_WAVES   = 4;
constexpr int SCAN_THREADS = SCAN_WAVES * 32;
constexpr int SCAN_BLOCKS  = BS / (SCAN_WAVES * 16);
constexpr size_t LDS_STATE_PER_WAVE = 24576;                        // bytes
constexpr size_t LDS_STATE = (size_t)SCAN_WAVES * LDS_STATE_PER_WAVE; // 96KB
// staged weights (f16 halves): fc2w | wq2 | wk2 | wq3 | wk3
constexpr size_t LW_FC2W = 0;
constexpr size_t LW_WQ2  = LW_FC2W + 256 * 128;
constexpr size_t LW_WK2  = LW_WQ2 + 128 * 128;
constexpr size_t LW_WQ3  = LW_WK2 + 128 * 128;
constexpr size_t LW_WK3  = LW_WQ3 + 128 * 128;
constexpr size_t LW_HALVES = LW_WK3 + 128 * 128;                    // 98304 halves
constexpr size_t LDS_TOTAL = LDS_STATE + LW_HALVES * 2;             // 288KB

DEV float sigmoidf(float x) { return 1.0f / (1.0f + __expf(-x)); }

DEV v8f wmma32(v16h a, v16h b, v8f c) {
  return __builtin_amdgcn_wmma_f32_16x16x32_f16(false, a, false, b, (short)0, c,
                                                false, false);
}

// B fragment: packed fragment-linear layout, 512 halves per 16x32 K-tile x
// 16 N-tile fragment, 16 contiguous halves (32B) per lane.
DEV v16h bfrag(const _Float16* __restrict__ wf, int frag, int lane) {
  return *(const v16h*)(wf + (size_t)frag * 512 + (size_t)lane * 16);
}

// A fragment (16x32 f16, M x K) from a per-lane f16 row pointer (row for m = lane&15).
// lanes 0-15: a[0..7]=K kb..kb+7, a[8..15]=K 16+kb..; kb = (lane>>4)*8
DEV v16h afrag_f16row(const _Float16* __restrict__ row, int kt, int lane) {
  int kb = (lane >> 4) * 8;
  const _Float16* p = row + kt * 32 + kb;
  v8h g0 = *(const v8h*)(p);
  v8h g1 = *(const v8h*)(p + 16);
  v16h a;
#pragma unroll
  for (int j = 0; j < 8; ++j) { a[j] = g0[j]; a[8 + j] = g1[j]; }
  return a;
}

// A fragment from a 16x128 f32 tile (LDS), row-major stride DIM
DEV v16h afrag_f32(const float* base, int kt, int lane) {
  int m = lane & 15, kb = (lane >> 4) * 8;
  const float* p = base + m * DIM + kt * 32 + kb;
  v4f x0 = *(const v4f*)(p);
  v4f x1 = *(const v4f*)(p + 4);
  v4f x2 = *(const v4f*)(p + 16);
  v4f x3 = *(const v4f*)(p + 20);
  v16h a;
#pragma unroll
  for (int j = 0; j < 4; ++j) {
    a[j]      = (_Float16)x0[j];
    a[4 + j]  = (_Float16)x1[j];
    a[8 + j]  = (_Float16)x2[j];
    a[12 + j] = (_Float16)x3[j];
  }
  return a;
}

// A fragment from a 16x128 f16 tile (LDS), row-major stride DIM
DEV v16h afrag_h16(const _Float16* base, int kt, int lane) {
  int m = lane & 15;
  return afrag_f16row(base + m * DIM, kt, lane);
}

DEV void zacc(v8f* acc) {
  v8f z = {0.f, 0.f, 0.f, 0.f, 0.f, 0.f, 0.f, 0.f};
#pragma unroll
  for (int nt = 0; nt < 8; ++nt) acc[nt] = z;
}

template <typename GetA>
DEV void gemm_k128(v8f* acc, GetA getA, const _Float16* __restrict__ wf, int lane) {
#pragma unroll
  for (int kt = 0; kt < 4; ++kt) {
    v16h a = getA(kt);
#pragma unroll
    for (int nt = 0; nt < 8; ++nt)
      acc[nt] = wmma32(a, bfrag(wf, kt * 8 + nt, lane), acc[nt]);
  }
}

// C/D layout: element r of acc[nt]: m = (lane>>4)*8 + r, n = nt*16 + (lane&15)
DEV void store_acc_f16(_Float16* dst, const v8f* acc, int lane) {
  int nl = lane & 15, rb = (lane >> 4) * 8;
#pragma unroll
  for (int nt = 0; nt < 8; ++nt)
#pragma unroll
    for (int r = 0; r < 8; ++r)
      dst[(rb + r) * DIM + nt * 16 + nl] = (_Float16)acc[nt][r];
}

// ---------------------------------------------------------------------------
// Prep kernels
// ---------------------------------------------------------------------------
__global__ void cvt_f16_kernel(const float* __restrict__ src,
                               _Float16* __restrict__ dst, int n) {
  int i = blockIdx.x * blockDim.x + threadIdx.x;
  if (i < n) dst[i] = (_Float16)src[i];
}

// Pack a (K x N) f32 weight (y = x @ W) into fragment-linear f16 B layout.
__global__ void pack_w_kernel(const float* __restrict__ W,
                              _Float16* __restrict__ out, int K, int N) {
  int tid = blockIdx.x * blockDim.x + threadIdx.x;
  if (tid >= K * N) return;
  int i    = tid & 15;
  int lane = (tid >> 4) & 31;
  int f    = tid >> 9;
  int n16  = N >> 4;
  int nt = f % n16, kt = f / n16;
  int k = kt * 32 + ((lane >> 4) * 8) + (i & 7) + ((i >> 3) * 16);
  int n = nt * 16 + (lane & 15);
  out[tid] = (_Float16)W[(size_t)k * N + n];
}

// x = f16( concat(skill_emb[skill], diff_emb[diff]) @ fc_W + fc_b ), all BS*L rows
__global__ __launch_bounds__(256) void xprep_kernel(
    _Float16* __restrict__ ws, const int* __restrict__ skill,
    const int* __restrict__ diff, const float* __restrict__ fcb) {
  const int lane = threadIdx.x & 31, wave = threadIdx.x >> 5;
  const int tile = blockIdx.x * 8 + wave;
  const int rr = tile * 16 + (lane & 15);
  const _Float16* skh = ws + OFF_SKH;
  const _Float16* dfh = ws + OFF_DFH;
  const _Float16* fcw = ws + OFF_FCW;
  _Float16* xh = ws + OFF_XH;
  const _Float16* skrow = skh + (size_t)skill[rr] * DIM;
  const _Float16* dfrow = dfh + (size_t)diff[rr] * DIM;
  v8f acc[8];
  zacc(acc);
#pragma unroll
  for (int kt = 0; kt < 8; ++kt) {
    v16h a = (kt < 4) ? afrag_f16row(skrow, kt, lane)
                      : afrag_f16row(dfrow, kt - 4, lane);
#pragma unroll
    for (int nt = 0; nt < 8; ++nt)
      acc[nt] = wmma32(a, bfrag(fcw, kt * 8 + nt, lane), acc[nt]);
  }
  const int nl = lane & 15, rb = (lane >> 4) * 8;
#pragma unroll
  for (int nt = 0; nt < 8; ++nt) {
    int n = nt * 16 + nl;
    float b = fcb[n];
#pragma unroll
    for (int r = 0; r < 8; ++r) {
      size_t row = (size_t)tile * 16 + rb + r;
      xh[row * DIM + n] = (_Float16)(acc[nt][r] + b);
    }
  }
}

// ---------------------------------------------------------------------------
// Sequential scan: 8 blocks x 4 waves, each wave owns 16 batch rows for all
// 199 steps. h/c master state f32 in LDS, staging tiles f16 in LDS, and the
// 5 hottest weights pinned in LDS (fragments served by ds_load instead of L2).
// ---------------------------------------------------------------------------
__global__ __launch_bounds__(SCAN_THREADS) void scan_kernel(
    _Float16* __restrict__ ws, const int* __restrict__ hints,
    const int* __restrict__ answer, const float* __restrict__ fc2b,
    const float* __restrict__ b11, const float* __restrict__ b12,
    const float* __restrict__ b21, const float* __restrict__ b22,
    float* __restrict__ out) {
  extern __shared__ char smem[];
  const int lane = threadIdx.x & 31, wave = threadIdx.x >> 5;
  const int b0 = blockIdx.x * (SCAN_WAVES * 16) + wave * 16;
  char* wl = smem + (size_t)wave * LDS_STATE_PER_WAVE;
  float*    hS  = (float*)(wl);             // 16x128 f32
  float*    cS  = (float*)(wl + 8192);      // 16x128 f32
  _Float16* xtS = (_Float16*)(wl + 16384);  // 16x128 f16
  _Float16* acS = (_Float16*)(wl + 20480);  // 16x128 f16 (k2/out/k3/o3 staging)
  _Float16* lw  = (_Float16*)(smem + LDS_STATE);

  // ---- stage hot weights into LDS (two contiguous workspace chunks)
  {
    const v4u* srcA = (const v4u*)(ws + OFF_FC2W);  // fc2w | wq2 | wk2
    v4u* dstA = (v4u*)(lw + LW_FC2W);
    for (int i = threadIdx.x; i < (int)((256 * 128 + 2 * 128 * 128) / 8);
         i += blockDim.x)
      dstA[i] = srcA[i];
    const v4u* srcB = (const v4u*)(ws + OFF_WQ3);   // wq3 | wk3
    v4u* dstB = (v4u*)(lw + LW_WQ3);
    for (int i = threadIdx.x; i < (int)((2 * 128 * 128) / 8); i += blockDim.x)
      dstB[i] = srcB[i];
  }

  const _Float16* fc2w = lw + LW_FC2W;
  const _Float16* wq2  = lw + LW_WQ2;
  const _Float16* wk2  = lw + LW_WK2;
  const _Float16* wq3  = lw + LW_WQ3;
  const _Float16* wk3  = lw + LW_WK3;
  const _Float16* wv2  = ws + OFF_WV2;
  const _Float16* wv3  = ws + OFF_WV3;
  const _Float16* f11w = ws + OFF_F11;
  const _Float16* f12w = ws + OFF_F12;
  const _Float16* f21w = ws + OFF_F21;
  const _Float16* f22w = ws + OFF_F22;
  const _Float16* xh   = ws + OFF_XH;
  const _Float16* anh  = ws + OFF_ANH;
  const _Float16* hnh  = ws + OFF_HNH;

  for (int i = lane; i < 2048; i += 32) { hS[i] = 0.f; cS[i] = 0.f; }
  __syncthreads();

  const float INV = 0.08838834764831845f;  // 1/sqrt(128)
  const int nl = lane & 15, rb = (lane >> 4) * 8;
  const int brow = b0 + nl;  // A-matrix row handled by this lane

  for (int t = 0; t < SEQL - 1; ++t) {
    // --- x_t = f16( concat(x[:,t], h) @ fc2_W + fc2_b ) -> xtS
    {
      v8f acc[8];
      zacc(acc);
      const _Float16* xrow = xh + ((size_t)brow * SEQL + t) * DIM;
#pragma unroll
      for (int kt = 0; kt < 8; ++kt) {
        v16h a = (kt < 4) ? afrag_f16row(xrow, kt, lane)
                          : afrag_f32(hS, kt - 4, lane);
#pragma unroll
        for (int nt = 0; nt < 8; ++nt)
          acc[nt] = wmma32(a, bfrag(fc2w, kt * 8 + nt, lane), acc[nt]);
      }
#pragma unroll
      for (int nt = 0; nt < 8; ++nt) {
        int n = nt * 16 + nl;
        float b = fc2b[n];
#pragma unroll
        for (int r = 0; r < 8; ++r)
          xtS[(rb + r) * DIM + n] = (_Float16)(acc[nt][r] + b);
      }
    }
    __syncthreads();

    // --- k2 = c @ Wk2 -> acS (raw f16)
    {
      v8f acc[8];
      zacc(acc);
      gemm_k128(acc, [&](int kt) { return afrag_f32(cS, kt, lane); }, wk2, lane);
      store_acc_f16(acS, acc, lane);
    }
    __syncthreads();

    // --- v2 = hints_emb[hints_t] @ Wv2 (regs); q2 = x_t @ Wq2;
    //     out = v2 * sigmoid(q2*k2*inv) -> acS
    {
      v8f accV[8];
      zacc(accV);
      const _Float16* hrow =
          hnh + (size_t)hints[(size_t)brow * SEQL + t] * DIM;
      gemm_k128(accV, [&](int kt) { return afrag_f16row(hrow, kt, lane); },
                wv2, lane);
      v8f accQ[8];
      zacc(accQ);
      gemm_k128(accQ, [&](int kt) { return afrag_h16(xtS, kt, lane); }, wq2,
                lane);
#pragma unroll
      for (int nt = 0; nt < 8; ++nt) {
        int n = nt * 16 + nl;
#pragma unroll
        for (int r = 0; r < 8; ++r) {
          int m = rb + r;
          float k2v = (float)acS[m * DIM + n];
          acS[m * DIM + n] =
              (_Float16)(accV[nt][r] * sigmoidf(accQ[nt][r] * k2v * INV));
        }
      }
    }
    __syncthreads();

    // --- c += tanh(out @ f11 + b11) * sigmoid(out @ f12 + b12)
    {
      v8f accT[8];
      zacc(accT);
      gemm_k128(accT, [&](int kt) { return afrag_h16(acS, kt, lane); }, f11w,
                lane);
      v8f accS[8];
      zacc(accS);
      gemm_k128(accS, [&](int kt) { return afrag_h16(acS, kt, lane); }, f12w,
                lane);
#pragma unroll
      for (int nt = 0; nt < 8; ++nt) {
        int n = nt * 16 + nl;
        float bb1 = b11[n], bb2 = b12[n];
#pragma unroll
        for (int r = 0; r < 8; ++r) {
          int m = rb + r;
          cS[m * DIM + n] +=
              tanhf(accT[nt][r] + bb1) * sigmoidf(accS[nt][r] + bb2);
        }
      }
    }
    __syncthreads();

    // --- k3 = h @ Wq3 -> acS (raw f16)   (h is still the old h)
    {
      v8f acc[8];
      zacc(acc);
      gemm_k128(acc, [&](int kt) { return afrag_f32(hS, kt, lane); }, wq3, lane);
      store_acc_f16(acS, acc, lane);
    }
    __syncthreads();

    // --- v3 = answer_emb[a_t] @ Wv3; q3 = x_t @ Wk3;
    //     o3 = v3 * sigmoid(q3*k3*inv) -> acS
    {
      v8f accV[8];
      zacc(accV);
      const _Float16* arow =
          anh + (size_t)answer[(size_t)brow * SEQL + t] * DIM;
      gemm_k128(accV, [&](int kt) { return afrag_f16row(arow, kt, lane); },
                wv3, lane);
      v8f accQ[8];
      zacc(accQ);
      gemm_k128(accQ, [&](int kt) { return afrag_h16(xtS, kt, lane); }, wk3,
                lane);
#pragma unroll
      for (int nt = 0; nt < 8; ++nt) {
        int n = nt * 16 + nl;
#pragma unroll
        for (int r = 0; r < 8; ++r) {
          int m = rb + r;
          float k3v = (float)acS[m * DIM + n];
          acS[m * DIM + n] =
              (_Float16)(accV[nt][r] * sigmoidf(accQ[nt][r] * k3v * INV));
        }
      }
    }
    __syncthreads();

    // --- h = tanh(o3 @ f21 + b21) * sigmoid(o3 @ f22 + b22) + c_new
    {
      v8f accT[8];
      zacc(accT);
      gemm_k128(accT, [&](int kt) { return afrag_h16(acS, kt, lane); }, f21w,
                lane);
      v8f accS[8];
      zacc(accS);
      gemm_k128(accS, [&](int kt) { return afrag_h16(acS, kt, lane); }, f22w,
                lane);
#pragma unroll
      for (int nt = 0; nt < 8; ++nt) {
        int n = nt * 16 + nl;
        float bb1 = b21[n], bb2 = b22[n];
#pragma unroll
        for (int r = 0; r < 8; ++r) {
          int m = rb + r;
          hS[m * DIM + n] = tanhf(accT[nt][r] + bb1) *
                                sigmoidf(accS[nt][r] + bb2) +
                            cS[m * DIM + n];
        }
      }
    }
    __syncthreads();

    // --- y[b,t] = sigmoid( dot(x[:,t+1], h_new) )
    if (lane < 16) {
      int b = b0 + lane;
      const _Float16* xn = xh + ((size_t)b * SEQL + (t + 1)) * DIM;
      float s = 0.f;
#pragma unroll 4
      for (int k = 0; k < DIM; k += 8) {
        v8h xv = *(const v8h*)(xn + k);
#pragma unroll
        for (int j = 0; j < 8; ++j)
          s += (float)xv[j] * hS[lane * DIM + k + j];
      }
      out[(size_t)b * SEQL + t] = sigmoidf(s);
    }
    if (t + 2 < SEQL)  // pull next step's activation row toward the WGP
      __builtin_prefetch(xh + ((size_t)brow * SEQL + (t + 2)) * DIM, 0, 1);
    __syncthreads();
  }
  if (lane < 16) out[(size_t)(b0 + lane) * SEQL + (SEQL - 1)] = 0.f;
}

// ---------------------------------------------------------------------------
extern "C" void kernel_launch(void* const* d_in, const int* in_sizes, int n_in,
                              void* d_out, int out_size, void* d_ws,
                              size_t ws_size, hipStream_t stream) {
  (void)in_sizes; (void)n_in; (void)out_size;
  if (ws_size < WS_HALVES * sizeof(_Float16)) return;
  _Float16* ws = (_Float16*)d_ws;
  auto g = [](int n) { return (n + 255) / 256; };

  // f16 copies of embedding tables
  cvt_f16_kernel<<<g(1002 * 128), 256, 0, stream>>>((const float*)d_in[5],
                                                    ws + OFF_SKH, 1002 * 128);
  cvt_f16_kernel<<<g(3 * 128), 256, 0, stream>>>((const float*)d_in[6],
                                                 ws + OFF_ANH, 3 * 128);
  cvt_f16_kernel<<<g(102 * 128), 256, 0, stream>>>((const float*)d_in[7],
                                                   ws + OFF_DFH, 102 * 128);
  cvt_f16_kernel<<<g(12 * 128), 256, 0, stream>>>((const float*)d_in[8],
                                                  ws + OFF_HNH, 12 * 128);

  // pack all weights into WMMA B-fragment layout (f16)
  pack_w_kernel<<<g(256 * 128), 256, 0, stream>>>((const float*)d_in[16],
                                                  ws + OFF_FCW, 256, 128);
  pack_w_kernel<<<g(256 * 128), 256, 0, stream>>>((const float*)d_in[18],
                                                  ws + OFF_FC2W, 256, 128);
  pack_w_kernel<<<g(128 * 128), 256, 0, stream>>>((const float*)d_in[10],
                                                  ws + OFF_WQ2, 128, 128);
  pack_w_kernel<<<g(128 * 128), 256, 0, stream>>>((const float*)d_in[11],
                                                  ws + OFF_WK2, 128, 128);
  pack_w_kernel<<<g(128 * 128), 256, 0, stream>>>((const float*)d_in[12],
                                                  ws + OFF_WV2, 128, 128);
  pack_w_kernel<<<g(128 * 128), 256, 0, stream>>>((const float*)d_in[13],
                                                  ws + OFF_WQ3, 128, 128);
  pack_w_kernel<<<g(128 * 128), 256, 0, stream>>>((const float*)d_in[14],
                                                  ws + OFF_WK3, 128, 128);
  pack_w_kernel<<<g(128 * 128), 256, 0, stream>>>((const float*)d_in[15],
                                                  ws + OFF_WV3, 128, 128);
  pack_w_kernel<<<g(128 * 128), 256, 0, stream>>>((const float*)d_in[20],
                                                  ws + OFF_F11, 128, 128);
  pack_w_kernel<<<g(128 * 128), 256, 0, stream>>>((const float*)d_in[22],
                                                  ws + OFF_F12, 128, 128);
  pack_w_kernel<<<g(128 * 128), 256, 0, stream>>>((const float*)d_in[24],
                                                  ws + OFF_F21, 128, 128);
  pack_w_kernel<<<g(128 * 128), 256, 0, stream>>>((const float*)d_in[26],
                                                  ws + OFF_F22, 128, 128);

  // prologue GEMM over all 102400 rows: 6400 M-tiles, 8 waves/block
  xprep_kernel<<<(BS * SEQL / 16) / 8, 256, 0, stream>>>(
      ws, (const int*)d_in[0], (const int*)d_in[2], (const float*)d_in[17]);

  // sequential scan: 8 blocks x 128 threads, 288KB dynamic LDS per block
  scan_kernel<<<SCAN_BLOCKS, SCAN_THREADS, LDS_TOTAL, stream>>>(
      ws, (const int*)d_in[3], (const int*)d_in[1], (const float*)d_in[19],
      (const float*)d_in[21], (const float*)d_in[23], (const float*)d_in[25],
      (const float*)d_in[27], (float*)d_out);
}
